// DeepseekV3MoE_33200097198620
// MI455X (gfx1250) — matrix operand order repaired and checked by reading, not compiled, not run on hardware
//
#include <hip/hip_runtime.h>
#include <hip/hip_bf16.h>
#include <cstdint>

// ---------------------------------------------------------------------------
// DeepSeek-V3 MoE block for MI455X (gfx1250, wave32, WMMA bf16)
//   T=2048 tokens, D=1024, E=8 routed experts top-2, F=1408, shared F2=2816
// Pipeline:
//   prep:   weights fp32 -> bf16, pre-swizzled into WMMA-B fragment lines
//   gemm:   async global->LDS copy of fragment lines, ds_load_b128, WMMA
// ---------------------------------------------------------------------------

#define TT     2048
#define DD     1024
#define FF     1408
#define FF2    2816
#define EE     8
#define SLOTS  4096   // T * top_k

typedef __bf16 bf16_t;
typedef bf16_t v16bf __attribute__((ext_vector_type(16)));
typedef float  v8f   __attribute__((ext_vector_type(8)));

union BF16x16 { uint4 q[2]; v16bf v; };

__device__ __forceinline__ unsigned short f2bf(float f) {
    unsigned u = __float_as_uint(f);
    u += 0x7fffu + ((u >> 16) & 1u);          // round-to-nearest-even
    return (unsigned short)(u >> 16);
}

// Low 32 bits of a flat pointer into the LDS aperture == byte offset in LDS.
__device__ __forceinline__ unsigned lds_off(const void* p) {
    return (unsigned)(uintptr_t)p;
}

// CDNA5 async copy: 16 bytes / lane, global -> LDS, tracked by ASYNCcnt.
__device__ __forceinline__ void async_copy16(unsigned ldsByteOff, const void* g) {
    unsigned long long ga = (unsigned long long)(uintptr_t)g;
    asm volatile("global_load_async_to_lds_b128 %0, %1, off"
                 :: "v"(ldsByteOff), "v"(ga) : "memory");
}
__device__ __forceinline__ void wait_async() {
    asm volatile("s_wait_asynccnt 0x0" ::: "memory");
}

// ---------------------------------------------------------------------------
// 1) fp32 -> bf16 conversion of activations (vectorized x4)
// ---------------------------------------------------------------------------
__global__ void cvt_bf16_kernel(const float* __restrict__ x,
                                unsigned short* __restrict__ xb, int n4) {
    int i = blockIdx.x * blockDim.x + threadIdx.x;
    if (i >= n4) return;
    float4 v = ((const float4*)x)[i];
    ushort4 o;
    o.x = f2bf(v.x); o.y = f2bf(v.y); o.z = f2bf(v.z); o.w = f2bf(v.w);
    ((ushort4*)xb)[i] = o;
}

// ---------------------------------------------------------------------------
// 2) Weight prep: fp32 [Kd, Nd] -> bf16 pre-swizzled WMMA-B fragment lines.
//    Line L = (n/16)*(Kd/32) + (k/32); within a line: 32 lanes x 16 bf16,
//    lane = (n&15) | (((k>>4)&1)<<4), ushort idx = k&15  (contiguous k order, same as A).
//    One thread produces one 32-byte lane-chunk (contiguous store).
// ---------------------------------------------------------------------------
__global__ __launch_bounds__(256)
void swz_kernel(const float* __restrict__ W, unsigned short* __restrict__ out,
                int Kd, int Nd, size_t eStride) {
    const float* Wp = W + (size_t)blockIdx.y * eStride;
    unsigned short* op = out + (size_t)blockIdx.y * eStride;
    const int gid = blockIdx.x * 256 + threadIdx.x;
    const int KT = Kd >> 5;
    const int total = (Nd >> 4) * KT * 32;
    if (gid >= total) return;
    const int lane = gid & 31;
    const int line = gid >> 5;
    const int kt   = line % KT;
    const int nsub = line / KT;
    const int n     = (nsub << 4) + (lane & 15);
    const int kbase = (kt << 5) + ((lane >> 4) << 4);
    union { unsigned short h[16]; uint4 q[2]; } u;
#pragma unroll
    for (int i = 0; i < 16; ++i)
        u.h[i] = f2bf(Wp[(size_t)(kbase + i) * Nd + n]);
    uint4* dst = (uint4*)(op + (size_t)gid * 16);
    dst[0] = u.q[0];
    dst[1] = u.q[1];
}

// ---------------------------------------------------------------------------
// 3) MoE gate: one wave32 per token. logits[E=8] -> softmax -> top2 ->
//    normalized * 2.5
// ---------------------------------------------------------------------------
__global__ __launch_bounds__(256)
void gate_topk_kernel(const float* __restrict__ x, const float* __restrict__ gw,
                      int* __restrict__ topki, float* __restrict__ topkw) {
    const int wave = threadIdx.x >> 5, lane = threadIdx.x & 31;
    const int t = blockIdx.x * 8 + wave;
    const float4* xr = (const float4*)(x + (size_t)t * DD);
    float acc[EE];
#pragma unroll
    for (int e = 0; e < EE; ++e) acc[e] = 0.f;
#pragma unroll
    for (int i = 0; i < 8; ++i) {
        float4 xv = xr[lane + i * 32];
#pragma unroll
        for (int e = 0; e < EE; ++e) {
            float4 gv = *(const float4*)(gw + (size_t)e * DD + (lane + i * 32) * 4);
            acc[e] += xv.x * gv.x + xv.y * gv.y + xv.z * gv.z + xv.w * gv.w;
        }
    }
#pragma unroll
    for (int e = 0; e < EE; ++e)
        for (int o = 16; o > 0; o >>= 1) acc[e] += __shfl_xor(acc[e], o, 32);
    if (lane == 0) {
        float mx = acc[0];
#pragma unroll
        for (int e = 1; e < EE; ++e) mx = fmaxf(mx, acc[e]);
        float p[EE], s = 0.f;
#pragma unroll
        for (int e = 0; e < EE; ++e) { p[e] = __expf(acc[e] - mx); s += p[e]; }
        float inv = __builtin_amdgcn_rcpf(s);
#pragma unroll
        for (int e = 0; e < EE; ++e) p[e] *= inv;
        int i0 = 0;
#pragma unroll
        for (int e = 1; e < EE; ++e) if (p[e] > p[i0]) i0 = e;
        int i1 = (i0 == 0) ? 1 : 0;
#pragma unroll
        for (int e = 0; e < EE; ++e) if (e != i0 && p[e] > p[i1]) i1 = e;
        float w0 = p[i0], w1 = p[i1];
        float sc = 2.5f * __builtin_amdgcn_rcpf(w0 + w1 + 1e-20f);
        topki[t * 2 + 0] = i0; topki[t * 2 + 1] = i1;
        topkw[t * 2 + 0] = w0 * sc; topkw[t * 2 + 1] = w1 * sc;
    }
}

// ---------------------------------------------------------------------------
// 4) Dispatch: counts -> exclusive scan -> compact slot lists (single block)
// ---------------------------------------------------------------------------
__global__ __launch_bounds__(256)
void dispatch_kernel(const int* __restrict__ topki, const float* __restrict__ topkw,
                     int* __restrict__ tok_of_slot, float* __restrict__ w_of_slot,
                     int* __restrict__ gbase, int* __restrict__ gcnt) {
    __shared__ int scnt[EE], scur[EE];
    const int tid = threadIdx.x;
    if (tid < EE) scnt[tid] = 0;
    __syncthreads();
    for (int i = tid; i < SLOTS; i += 256) atomicAdd(&scnt[topki[i]], 1);
    __syncthreads();
    if (tid == 0) {
        int b = 0;
        for (int e = 0; e < EE; ++e) { int c = scnt[e]; gbase[e] = b; gcnt[e] = c; scur[e] = b; b += c; }
    }
    __syncthreads();
    for (int i = tid; i < SLOTS; i += 256) {
        int e = topki[i];
        int s = atomicAdd(&scur[e], 1);
        tok_of_slot[s] = i >> 1;
        w_of_slot[s]   = topkw[i];
    }
}

// ---------------------------------------------------------------------------
// 5) Fused gate+up GEMM on pre-swizzled bf16 weights.
//    8 waves tile 64 rows x 64 cols; K over D in steps of 32.
//    Per K-step: async-copy 8 fragment lines (2 mats x 4 subtiles, 1KB each)
//    into LDS, then ds_load_b128 fragments + 4 WMMAs per wave.
// ---------------------------------------------------------------------------
template <bool ROUTED>
__global__ __launch_bounds__(256)
void gateup_kernel(const unsigned short* __restrict__ xb,
                   const unsigned short* __restrict__ swg,
                   const unsigned short* __restrict__ swu,
                   int Fdim, unsigned short* __restrict__ Hout,
                   const int* __restrict__ tok_of_slot,
                   const int* __restrict__ gbase, const int* __restrict__ gcnt) {
    constexpr int Kdim = DD;
    constexpr int KT = Kdim / 32;
    __shared__ unsigned short frag[2][4][32][16];   // 8 KB

    const int e = blockIdx.z;
    int m0, mlimit;
    const unsigned short* Wm[2];
    if constexpr (ROUTED) {
        const int cnt = gcnt[e];
        if ((int)blockIdx.y * 64 >= cnt) return;
        const int base = gbase[e];
        m0 = base + blockIdx.y * 64;
        mlimit = base + cnt;
        const size_t wo = (size_t)e * Kdim * Fdim;
        Wm[0] = swg + wo; Wm[1] = swu + wo;
    } else {
        m0 = blockIdx.y * 64; mlimit = TT;
        Wm[0] = swg; Wm[1] = swu;
    }
    const int n0   = blockIdx.x * 64;
    const int tid  = threadIdx.x, lane = tid & 31, wave = tid >> 5;
    const int wm   = wave >> 1, wn = wave & 1;

    const int rloc = wm * 16 + (lane & 15);
    int srow = m0 + rloc; if (srow > mlimit - 1) srow = mlimit - 1;
    const int tok = ROUTED ? tok_of_slot[srow] : srow;
    const unsigned short* arow = xb + (size_t)tok * Kdim;
    const int aoff = (lane >> 4) << 4;              // 0 or 16

    v8f acc[2][2];
#pragma unroll
    for (int m = 0; m < 2; ++m)
#pragma unroll
        for (int s = 0; s < 2; ++s)
#pragma unroll
            for (int i = 0; i < 8; ++i) acc[m][s][i] = 0.f;

    for (int kt = 0; kt < KT; ++kt) {
        // stage 8 fragment lines (2 matrices x 4 subtiles): 512 chunks of 16B
#pragma unroll
        for (int j = 0; j < 2; ++j) {
            const int c   = tid + 256 * j;
            const int m   = c >> 8;
            const int sub = (c >> 6) & 3;
            const int ch  = c & 63;
            const size_t line = (size_t)((n0 >> 4) + sub) * KT + kt;
            async_copy16(lds_off(&frag[0][0][0][0] + (size_t)c * 8),
                         Wm[m] + line * 512 + (size_t)ch * 8);
        }
        wait_async();
        __syncthreads();

        const int kk = kt * 32;
        BF16x16 a;
        a.q[0] = *(const uint4*)(arow + kk + aoff);
        a.q[1] = *(const uint4*)(arow + kk + aoff + 8);
#pragma unroll
        for (int m = 0; m < 2; ++m)
#pragma unroll
            for (int s = 0; s < 2; ++s) {
                BF16x16 b;
                const uint4* fp = (const uint4*)&frag[m][wn * 2 + s][lane][0];
                b.q[0] = fp[0]; b.q[1] = fp[1];
                acc[m][s] = __builtin_amdgcn_wmma_f32_16x16x32_bf16(
                    false, a.v, false, b.v, (short)0, acc[m][s], false, false);
            }
        __syncthreads();
    }

    // epilogue: h = silu(gate) * up  (fast rcp, no IEEE divide)
#pragma unroll
    for (int s = 0; s < 2; ++s)
#pragma unroll
        for (int r = 0; r < 8; ++r) {
            int mlocal = wm * 16 + r + ((lane >> 4) << 3);
            int slot = m0 + mlocal;
            if (slot < mlimit) {
                float g = acc[0][s][r], u = acc[1][s][r];
                float h = g * __builtin_amdgcn_rcpf(1.f + __expf(-g)) * u;
                int col = n0 + wn * 32 + s * 16 + (lane & 15);
                Hout[(size_t)slot * Fdim + col] = f2bf(h);
            }
        }
}

// ---------------------------------------------------------------------------
// 6) Down-projection GEMM on pre-swizzled bf16 weights.
//    ROUTED: out += combine_w * (H @ Wd) via f32 atomics
//    shared: out  = H_sh @ sw_down (plain store; initializes d_out)
// ---------------------------------------------------------------------------
template <bool ROUTED>
__global__ __launch_bounds__(256)
void down_kernel(const unsigned short* __restrict__ Hin,
                 const unsigned short* __restrict__ swd,
                 int Kdim, float* __restrict__ out,
                 const int* __restrict__ tok_of_slot, const float* __restrict__ w_of_slot,
                 const int* __restrict__ gbase, const int* __restrict__ gcnt) {
    constexpr int Dout = DD;
    __shared__ unsigned short frag[4][32][16];      // 4 KB
    const int KT = Kdim >> 5;

    const int e = blockIdx.z;
    int m0, mlimit;
    const unsigned short* W;
    if constexpr (ROUTED) {
        const int cnt = gcnt[e];
        if ((int)blockIdx.y * 64 >= cnt) return;
        const int base = gbase[e];
        m0 = base + blockIdx.y * 64;
        mlimit = base + cnt;
        W = swd + (size_t)e * Kdim * Dout;
    } else {
        m0 = blockIdx.y * 64; mlimit = TT; W = swd;
    }
    const int n0  = blockIdx.x * 64;
    const int tid = threadIdx.x, lane = tid & 31, wave = tid >> 5;
    const int wm  = wave >> 1, wn = wave & 1;

    const int rloc = wm * 16 + (lane & 15);
    int srow = m0 + rloc; if (srow > mlimit - 1) srow = mlimit - 1;
    const unsigned short* arow = Hin + (size_t)srow * Kdim;
    const int aoff = (lane >> 4) << 4;

    v8f acc[2];
#pragma unroll
    for (int s = 0; s < 2; ++s)
#pragma unroll
        for (int i = 0; i < 8; ++i) acc[s][i] = 0.f;

    for (int kt = 0; kt < KT; ++kt) {
        // 4 subtile lines: 256 chunks of 16B, one per thread
        {
            const int c   = tid;
            const int sub = (c >> 6) & 3;
            const int ch  = c & 63;
            const size_t line = (size_t)((n0 >> 4) + sub) * KT + kt;
            async_copy16(lds_off(&frag[0][0][0] + (size_t)c * 8),
                         W + line * 512 + (size_t)ch * 8);
        }
        wait_async();
        __syncthreads();

        const int kk = kt * 32;
        BF16x16 a;
        a.q[0] = *(const uint4*)(arow + kk + aoff);
        a.q[1] = *(const uint4*)(arow + kk + aoff + 8);
#pragma unroll
        for (int s = 0; s < 2; ++s) {
            BF16x16 b;
            const uint4* fp = (const uint4*)&frag[wn * 2 + s][lane][0];
            b.q[0] = fp[0]; b.q[1] = fp[1];
            acc[s] = __builtin_amdgcn_wmma_f32_16x16x32_bf16(
                false, a.v, false, b.v, (short)0, acc[s], false, false);
        }
        __syncthreads();
    }

#pragma unroll
    for (int r = 0; r < 8; ++r) {
        int mlocal = wm * 16 + r + ((lane >> 4) << 3);
        int slot = m0 + mlocal;
        if (slot < mlimit) {
            if constexpr (ROUTED) {
                int tk = tok_of_slot[slot];
                float w = w_of_slot[slot];
#pragma unroll
                for (int s = 0; s < 2; ++s) {
                    int col = n0 + wn * 32 + s * 16 + (lane & 15);
                    atomicAdd(&out[(size_t)tk * Dout + col], acc[s][r] * w);
                }
            } else {
#pragma unroll
                for (int s = 0; s < 2; ++s) {
                    int col = n0 + wn * 32 + s * 16 + (lane & 15);
                    out[(size_t)slot * Dout + col] = acc[s][r];
                }
            }
        }
    }
}

// ---------------------------------------------------------------------------
// Host launcher
// ---------------------------------------------------------------------------
extern "C" void kernel_launch(void* const* d_in, const int* in_sizes, int n_in,
                              void* d_out, int out_size, void* d_ws, size_t ws_size,
                              hipStream_t stream) {
    (void)in_sizes; (void)n_in; (void)out_size; (void)ws_size;

    const float* x       = (const float*)d_in[0];   // [T, D]
    const float* gate_w  = (const float*)d_in[1];   // [E, D]
    const float* w_gate  = (const float*)d_in[2];   // [E, D, F]
    const float* w_up    = (const float*)d_in[3];   // [E, D, F]
    const float* w_down  = (const float*)d_in[4];   // [E, F, D]
    const float* sw_gate = (const float*)d_in[5];   // [D, F2]
    const float* sw_up   = (const float*)d_in[6];   // [D, F2]
    const float* sw_down = (const float*)d_in[7];   // [F2, D]
    float* out = (float*)d_out;                     // [T, D]

    char* ws = (char*)d_ws;
    size_t off = 0;
    auto alloc = [&](size_t bytes) -> void* {
        void* p = ws + off;
        off = (off + bytes + 255) & ~(size_t)255;
        return p;
    };
    unsigned short* xb     = (unsigned short*)alloc((size_t)TT * DD * 2);
    unsigned short* h_rout = (unsigned short*)alloc((size_t)SLOTS * FF * 2);
    unsigned short* h_sh   = (unsigned short*)alloc((size_t)TT * FF2 * 2);
    unsigned short* swz_wg = (unsigned short*)alloc((size_t)EE * DD * FF * 2);
    unsigned short* swz_wu = (unsigned short*)alloc((size_t)EE * DD * FF * 2);
    unsigned short* swz_wd = (unsigned short*)alloc((size_t)EE * FF * DD * 2);
    unsigned short* swz_sg = (unsigned short*)alloc((size_t)DD * FF2 * 2);
    unsigned short* swz_su = (unsigned short*)alloc((size_t)DD * FF2 * 2);
    unsigned short* swz_sd = (unsigned short*)alloc((size_t)FF2 * DD * 2);
    int*   topki       = (int*)  alloc((size_t)TT * 2 * 4);
    float* topkw       = (float*)alloc((size_t)TT * 2 * 4);
    int*   tok_of_slot = (int*)  alloc((size_t)SLOTS * 4);
    float* w_of_slot   = (float*)alloc((size_t)SLOTS * 4);
    int*   gbase       = (int*)  alloc(EE * 4);
    int*   gcnt        = (int*)  alloc(EE * 4);

    // --- prep: activations + weights -> bf16 (weights pre-swizzled) ---
    cvt_bf16_kernel<<<(TT * DD / 4 + 255) / 256, 256, 0, stream>>>(x, xb, TT * DD / 4);
    {
        const int thrDF  = DD * FF  / 16;   // threads per expert matrix [D,F]
        const int thrFD  = FF * DD  / 16;
        const int thrDF2 = DD * FF2 / 16;
        const int thrF2D = FF2 * DD / 16;
        swz_kernel<<<dim3((thrDF  + 255) / 256, EE), 256, 0, stream>>>(w_gate,  swz_wg, DD,  FF, (size_t)DD * FF);
        swz_kernel<<<dim3((thrDF  + 255) / 256, EE), 256, 0, stream>>>(w_up,    swz_wu, DD,  FF, (size_t)DD * FF);
        swz_kernel<<<dim3((thrFD  + 255) / 256, EE), 256, 0, stream>>>(w_down,  swz_wd, FF,  DD, (size_t)FF * DD);
        swz_kernel<<<dim3((thrDF2 + 255) / 256, 1),  256, 0, stream>>>(sw_gate, swz_sg, DD,  FF2, 0);
        swz_kernel<<<dim3((thrDF2 + 255) / 256, 1),  256, 0, stream>>>(sw_up,   swz_su, DD,  FF2, 0);
        swz_kernel<<<dim3((thrF2D + 255) / 256, 1),  256, 0, stream>>>(sw_down, swz_sd, FF2, DD, 0);
    }
    // --- routing ---
    gate_topk_kernel<<<TT / 8, 256, 0, stream>>>(x, gate_w, topki, topkw);
    dispatch_kernel<<<1, 256, 0, stream>>>(topki, topkw, tok_of_slot, w_of_slot, gbase, gcnt);
    // --- shared expert (down initializes d_out) ---
    gateup_kernel<false><<<dim3(FF2 / 64, TT / 64, 1), 256, 0, stream>>>(
        xb, swz_sg, swz_su, FF2, h_sh, nullptr, nullptr, nullptr);
    down_kernel<false><<<dim3(DD / 64, TT / 64, 1), 256, 0, stream>>>(
        h_sh, swz_sd, FF2, out, nullptr, nullptr, nullptr, nullptr);
    // --- routed experts (grouped GEMMs, combine-weight atomics) ---
    gateup_kernel<true><<<dim3(FF / 64, TT / 64, EE), 256, 0, stream>>>(
        xb, swz_wg, swz_wu, FF, h_rout, tok_of_slot, gbase, gcnt);
    down_kernel<true><<<dim3(DD / 64, TT / 64, EE), 256, 0, stream>>>(
        h_rout, swz_wd, FF, out, tok_of_slot, w_of_slot, gbase, gcnt);
}
